// SuperPoint_86406152061196
// MI455X (gfx1250) — compile-verified, matrix operand order checked by reading, not run on hardware
//
#include <hip/hip_runtime.h>
#include <math.h>
#include <stdint.h>

typedef __attribute__((ext_vector_type(16))) _Float16 v16h;
typedef __attribute__((ext_vector_type(8)))  _Float16 v8h;
typedef __attribute__((ext_vector_type(8)))  float    v8f;
typedef __attribute__((ext_vector_type(4)))  int      v4i;

#define BB 2
#define HH 480
#define WW 640
#define HC 60
#define WC 80
#define DDIM 256
#define KTOP 1024
#define THRS 0.005f

#ifndef __has_builtin
#define __has_builtin(x) 0
#endif
#if __has_builtin(__builtin_amdgcn_global_load_async_to_lds_b128)
#define HAVE_ASYNC 1
#else
#define HAVE_ASYNC 0
#endif

// 16-byte global->LDS copy: async DMA path on CDNA5, sync fallback otherwise.
__device__ __forceinline__ void async_cp16(const _Float16* g, _Float16* l) {
#if HAVE_ASYNC
    __builtin_amdgcn_global_load_async_to_lds_b128(
        (__attribute__((address_space(1))) v4i*)(uintptr_t)g,
        (__attribute__((address_space(3))) v4i*)(uintptr_t)l, 0, 0);
#else
    *(v8h*)l = *(const v8h*)g;
#endif
}
__device__ __forceinline__ void wait_async() {
#if HAVE_ASYNC
#if __has_builtin(__builtin_amdgcn_s_wait_asynccnt)
    __builtin_amdgcn_s_wait_asynccnt(0);
#else
    asm volatile("s_wait_asynccnt 0" ::: "memory");
#endif
#endif
}

// All activations live in NHWC ([B][H][W][C]) f16. Weights are pre-transposed
// to [Cout][tap][Cin] f16 so WMMA fragments are contiguous 16B LDS loads.

// ------------------------------------------- weight convert + transpose (f32->f16)
__global__ void cvt_w_kernel(const float* __restrict__ src, _Float16* __restrict__ dst,
                             int Cout, int Cin, int taps) {
    int i = blockIdx.x * 256 + threadIdx.x;
    int total = Cout * Cin * taps;
    if (i >= total) return;
    int tap = i % taps; int t = i / taps;
    int ci = t % Cin;   int co = t / Cin;
    dst[((size_t)co * taps + tap) * Cin + ci] = (_Float16)src[i];
}

// ------------------------------------------------------- conv1a (Cin=1) direct
__global__ void conv1a_kernel(const float* __restrict__ img, const float* __restrict__ w,
                              const float* __restrict__ bias, _Float16* __restrict__ out,
                              int H, int W, int nb) {
    int idx = blockIdx.x * 256 + threadIdx.x;
    int total = nb * H * W;
    if (idx >= total) return;
    int x = idx % W; int t = idx / W;
    int y = t % H;   int b = t / H;
    float pix[9];
#pragma unroll
    for (int ky = 0; ky < 3; ++ky)
#pragma unroll
        for (int kx = 0; kx < 3; ++kx) {
            int gy = y + ky - 1, gx = x + kx - 1;
            pix[ky * 3 + kx] = (gy >= 0 && gy < H && gx >= 0 && gx < W)
                                   ? img[(b * H + gy) * W + gx] : 0.f;
        }
    _Float16* op = out + (size_t)idx * 64;
#pragma unroll
    for (int g = 0; g < 8; ++g) {
        v8h o;
#pragma unroll
        for (int r = 0; r < 8; ++r) {
            int co = g * 8 + r;
            float s = bias[co];
#pragma unroll
            for (int k = 0; k < 9; ++k) s += pix[k] * w[co * 9 + k];
            o[r] = (_Float16)fmaxf(s, 0.f);
        }
        *(v8h*)&op[g * 8] = o;
    }
}

// --------------------------------------------- implicit-GEMM WMMA conv (f16, NHWC)
// Block tile: 64 Cout x (8x8 pixels). 8 waves: wm in [0,4) -> 16 Cout rows,
// wn in [0,2) -> 32 pixels (two 16x16 f32 accumulators per wave).
// K is tap-major (k = tap*32 + ci, 32-channel chunks); LDS double-buffered with
// async global->LDS staging pipelined against the WMMA inner loop.
__device__ __forceinline__ v16h ldfrag(const _Float16* p, int half) {
    v8h lo = *(const v8h*)(p + half * 8);
    v8h hi = *(const v8h*)(p + 16 + half * 8);
    return __builtin_shufflevector(lo, hi, 0, 1, 2, 3, 4, 5, 6, 7,
                                   8, 9, 10, 11, 12, 13, 14, 15);
}

template <int KS>
__global__ __launch_bounds__(256)
void conv_wmma_kernel(const _Float16* __restrict__ in, const _Float16* __restrict__ wt,
                      const float* __restrict__ bias, _Float16* __restrict__ out,
                      int Cin, int Cout, int H, int W, int relu) {
    constexpr int TAPS    = KS * KS;
    constexpr int PAD     = KS >> 1;
    constexpr int HALO    = 8 + KS - 1;        // 10 (3x3) or 8 (1x1)
    constexpr int CHUNKK  = 32 * TAPS;         // K per 32-channel chunk
    constexpr int WSTRIDE = CHUNKK + 8;        // pad: conflict-free b128 column reads
    constexpr int SLOT    = 40;                // 32 ci + 8 pad halves per pixel

    const int tilesY = (H + 7) >> 3;
    const int tx = blockIdx.x;
    const int ty = blockIdx.y % tilesY;
    const int b  = blockIdx.y / tilesY;
    const int coBase = blockIdx.z * 64;

    const int tid  = threadIdx.x;
    const int lane = tid & 31;
    const int wave = tid >> 5;
    const int wm = wave & 3;
    const int wn = wave >> 2;

    __shared__ _Float16 ldsW[2][64 * WSTRIDE];
    __shared__ _Float16 ldsI[2][HALO * HALO * SLOT];
    __shared__ float    ldsB[64];

    if (tid < 64) {
        int co = coBase + tid;
        ldsB[tid] = (co < Cout) ? bias[co] : 0.f;
    }

    auto stage = [&](int bufi, int ci0) {
        // weights chunk: ldsW[co_l][tap*32+ci_l], 16B per thread-iter
        for (int i = tid; i < 64 * (CHUNKK / 8); i += 256) {
            int co_l = i / (CHUNKK / 8);
            int g    = i % (CHUNKK / 8);
            int tap  = g >> 2;
            int c8   = g & 3;
            int co   = coBase + co_l;
            _Float16* dst = &ldsW[bufi][co_l * WSTRIDE + tap * 32 + c8 * 8];
            if (co < Cout)
                async_cp16(&wt[((size_t)co * TAPS + tap) * Cin + ci0 + c8 * 8], dst);
            else { v8h z = {}; *(v8h*)dst = z; }
        }
        // activation halo: ldsI[pixel][ci], zero padded borders
        for (int i = tid; i < HALO * HALO * 4; i += 256) {
            int g   = i & 3;
            int pix = i >> 2;
            int iy  = pix / HALO, ix = pix % HALO;
            int gy = (ty << 3) + iy - PAD;
            int gx = (tx << 3) + ix - PAD;
            _Float16* dst = &ldsI[bufi][pix * SLOT + g * 8];
            if (gy >= 0 && gy < H && gx >= 0 && gx < W)
                async_cp16(&in[((size_t)(b * H + gy) * W + gx) * Cin + ci0 + g * 8], dst);
            else { v8h z = {}; *(v8h*)dst = z; }
        }
    };

    v8f acc0 = {}; v8f acc1 = {};

    const int nCh = Cin >> 5;
    stage(0, 0);
    wait_async();
    __syncthreads();

    const int mrow = lane & 15;
    const int half = lane >> 4;
    const int p0 = (wn << 5) + mrow;
    const int p1 = p0 + 16;

    for (int c = 0; c < nCh; ++c) {
        const int cur = c & 1;
        if (c + 1 < nCh) stage(cur ^ 1, (c + 1) << 5);   // prefetch next chunk

        const _Float16* lw = ldsW[cur];
        const _Float16* li = ldsI[cur];
#pragma unroll
        for (int s = 0; s < TAPS; ++s) {
            const int ky = s / KS, kx = s % KS;
            v16h aF  = ldfrag(&lw[(wm * 16 + mrow) * WSTRIDE + s * 32], half);
            v16h b0F = ldfrag(&li[(((p0 >> 3) + ky) * HALO + (p0 & 7) + kx) * SLOT], half);
            v16h b1F = ldfrag(&li[(((p1 >> 3) + ky) * HALO + (p1 & 7) + kx) * SLOT], half);
            acc0 = __builtin_amdgcn_wmma_f32_16x16x32_f16(false, aF, false, b0F,
                                                          (short)0, acc0, false, false);
            acc1 = __builtin_amdgcn_wmma_f32_16x16x32_f16(false, aF, false, b1F,
                                                          (short)0, acc1, false, false);
        }
        wait_async();
        __syncthreads();
    }

    // ---- epilogue: bias + ReLU, one b128 store of 8 consecutive Cout per acc
    const int n   = lane & 15;
    const int hm  = (lane >> 4) << 3;
    const int coB = coBase + wm * 16 + hm;        // 8 consecutive channels
    const bool vec = (coB + 8 <= Cout) && ((Cout & 7) == 0);
#pragma unroll
    for (int a = 0; a < 2; ++a) {
        const int p  = (wn << 5) + (a << 4) + n;
        const int gy = (ty << 3) + (p >> 3);
        const int gx = (tx << 3) + (p & 7);
        if (gy < H && gx < W) {
            _Float16* op = out + ((size_t)(b * H + gy) * W + gx) * Cout;
            if (vec) {
                v8h o;
#pragma unroll
                for (int r = 0; r < 8; ++r) {
                    float v = (a == 0 ? acc0[r] : acc1[r]) + ldsB[wm * 16 + hm + r];
                    if (relu && v < 0.f) v = 0.f;
                    o[r] = (_Float16)v;
                }
                *(v8h*)&op[coB] = o;
            } else {
#pragma unroll
                for (int r = 0; r < 8; ++r) {
                    int co = coB + r;
                    if (co < Cout) {
                        float v = (a == 0 ? acc0[r] : acc1[r]) + ldsB[wm * 16 + hm + r];
                        if (relu && v < 0.f) v = 0.f;
                        op[co] = (_Float16)v;
                    }
                }
            }
        }
    }
}

// ------------------------------------------------------------- 2x2 max pool (NHWC)
__global__ void pool2_kernel(const _Float16* __restrict__ in, _Float16* __restrict__ out,
                             int C, int H, int W, int nb) {
    int Ho = H >> 1, Wo = W >> 1;
    int idx = blockIdx.x * 256 + threadIdx.x;
    int total = nb * C * Ho * Wo;
    if (idx >= total) return;
    int c = idx % C; int t = idx / C;
    int x = t % Wo;  t /= Wo;
    int y = t % Ho;  int b = t / Ho;
    const _Float16* p = in + ((size_t)(b * H + 2 * y) * W + 2 * x) * C + c;
    size_t sC = C, sW = (size_t)W * C;
    float m = fmaxf(fmaxf((float)p[0], (float)p[sC]), fmaxf((float)p[sW], (float)p[sW + sC]));
    out[((size_t)(b * Ho + y) * Wo + x) * C + c] = (_Float16)m;
}

// ------------------------------- softmax over 65ch + pixel-shuffle to full res
__global__ void softmax_shuffle_kernel(const _Float16* __restrict__ pb, float* __restrict__ s0,
                                       int Hc, int Wc, int nb) {
    int idx = blockIdx.x * 256 + threadIdx.x;
    int total = nb * Hc * Wc;
    if (idx >= total) return;
    int wc = idx % Wc; int t = idx / Wc;
    int hc = t % Hc;   int b = t / Hc;
    const _Float16* p = pb + (size_t)idx * 65;
    float v[65]; float mx = -1e30f;
    for (int c = 0; c < 65; ++c) { v[c] = (float)p[c]; mx = fmaxf(mx, v[c]); }
    float sum = 0.f;
    for (int c = 0; c < 65; ++c) { v[c] = __expf(v[c] - mx); sum += v[c]; }
    float inv = 1.f / sum;
    const int Hf = Hc * 8, Wf = Wc * 8;
    for (int c = 0; c < 64; ++c)
        s0[((size_t)b * Hf + hc * 8 + (c >> 3)) * Wf + wc * 8 + (c & 7)] = v[c] * inv;
}

// ------------------------------------ separable 9x9 max pool (-inf padding)
__global__ void mp9h_kernel(const float* __restrict__ in, float* __restrict__ out,
                            int H, int W, int nb) {
    int idx = blockIdx.x * 256 + threadIdx.x;
    if (idx >= nb * H * W) return;
    int x = idx % W; int t = idx / W; int y = t % H; int b = t / H;
    const float* row = in + ((size_t)b * H + y) * W;
    float m = -INFINITY;
    for (int d = -4; d <= 4; ++d) { int xx = x + d; if (xx >= 0 && xx < W) m = fmaxf(m, row[xx]); }
    out[idx] = m;
}
__global__ void mp9v_kernel(const float* __restrict__ in, float* __restrict__ out,
                            int H, int W, int nb) {
    int idx = blockIdx.x * 256 + threadIdx.x;
    if (idx >= nb * H * W) return;
    int x = idx % W; int t = idx / W; int y = t % H; int b = t / H;
    float m = -INFINITY;
    for (int d = -4; d <= 4; ++d) { int yy = y + d; if (yy >= 0 && yy < H) m = fmaxf(m, in[((size_t)b * H + yy) * W + x]); }
    out[idx] = m;
}

// ----------------------------------------------------------- NMS elementwise
__global__ void eqmask_kernel(const float* s0, const float* pooled, float* m, int n) {
    int i = blockIdx.x * 256 + threadIdx.x;
    if (i < n) m[i] = (s0[i] == pooled[i]) ? 1.f : 0.f;
}
__global__ void suppress_kernel(const float* s0, const float* pmask, float* s3, int n) {
    int i = blockIdx.x * 256 + threadIdx.x;
    if (i < n) s3[i] = (pmask[i] > 0.f) ? 0.f : s0[i];
}
__global__ void update_kernel(float* m, const float* s3, const float* ps3, const float* pmask, int n) {
    int i = blockIdx.x * 256 + threadIdx.x;
    if (i < n) {
        bool newmax = (s3[i] == ps3[i]) && !(pmask[i] > 0.f);
        m[i] = ((m[i] > 0.f) || newmax) ? 1.f : 0.f;
    }
}
__global__ void finalmask_kernel(const float* s0, const float* m, float* s1, int n) {
    int i = blockIdx.x * 256 + threadIdx.x;
    if (i < n) s1[i] = ((m[i] > 0.f) && (s0[i] > THRS)) ? s0[i] : 0.f;
}

// ---------------------------------- top-K=1024 radix select + bitonic sort
__global__ __launch_bounds__(1024)
void topk_kernel(const float* __restrict__ masked, float* __restrict__ coords,
                 float* __restrict__ scores, int H, int W) {
    const int b = blockIdx.x, tid = threadIdx.x;
    const int N = H * W;
    const float* base = masked + (size_t)b * N;

    __shared__ unsigned hist[256];
    __shared__ unsigned sPrefix, sKneed, sCount;
    __shared__ float sv[KTOP];
    __shared__ int   si[KTOP];

    if (tid == 0) { sPrefix = 0; sKneed = KTOP; }
    __syncthreads();

    for (int shift = 24; shift >= 0; shift -= 8) {
        if (tid < 256) hist[tid] = 0;
        __syncthreads();
        unsigned pf = sPrefix;
        for (int i = tid; i < N; i += 1024) {
            unsigned bits = __float_as_uint(base[i]);
            bool cand = (shift == 24) || ((bits >> (shift + 8)) == pf);
            if (cand) atomicAdd(&hist[(bits >> shift) & 255u], 1u);
        }
        __syncthreads();
        if (tid == 0) {
            unsigned kn = sKneed, cum = 0; int bsel = 0;
            for (int bin = 255; bin >= 0; --bin) {
                if (cum + hist[bin] >= kn) { bsel = bin; break; }
                cum += hist[bin];
            }
            sKneed  = kn - cum;
            sPrefix = (sPrefix << 8) | (unsigned)bsel;
        }
        __syncthreads();
    }
    const unsigned thr = sPrefix;

    sv[tid] = 0.f; si[tid] = 0;
    if (tid == 0) sCount = 0;
    __syncthreads();
    for (int i = tid; i < N; i += 1024) {
        unsigned bits = __float_as_uint(base[i]);
        if (bits > thr) { unsigned p = atomicAdd(&sCount, 1u); if (p < KTOP) { sv[p] = base[i]; si[p] = i; } }
    }
    __syncthreads();
    for (int i = tid; i < N; i += 1024) {
        unsigned bits = __float_as_uint(base[i]);
        if (bits == thr) { unsigned p = atomicAdd(&sCount, 1u); if (p < KTOP) { sv[p] = base[i]; si[p] = i; } }
    }
    __syncthreads();

    for (int kk = 2; kk <= KTOP; kk <<= 1)
        for (int j = kk >> 1; j > 0; j >>= 1) {
            __syncthreads();
            int ixj = tid ^ j;
            if (ixj > tid) {
                bool desc = ((tid & kk) == 0);
                float v0 = sv[tid], v1 = sv[ixj];
                bool swp = desc ? (v0 < v1) : (v0 > v1);
                if (swp) {
                    sv[tid] = v1; sv[ixj] = v0;
                    int t = si[tid]; si[tid] = si[ixj]; si[ixj] = t;
                }
            }
        }
    __syncthreads();

    int idx = si[tid];
    coords[((size_t)b * KTOP + tid) * 2 + 0] = (float)(idx % W);
    coords[((size_t)b * KTOP + tid) * 2 + 1] = (float)(idx / W);
    scores[(size_t)b * KTOP + tid] = sv[tid];
}

// ------------------------------- descriptor map per-pixel L2 norm (NHWC in/out)
__global__ __launch_bounds__(256)
void l2norm_map_kernel(const _Float16* __restrict__ src, float* __restrict__ dmap,
                       int Hc, int Wc) {
    int pix = blockIdx.x;                  // (b*Hc + hc)*Wc + wc
    int d = threadIdx.x;
    float v = (float)src[(size_t)pix * DDIM + d];
    __shared__ float red[256];
    red[d] = v * v;
    __syncthreads();
    for (int s = 128; s > 0; s >>= 1) { if (d < s) red[d] += red[d + s]; __syncthreads(); }
    float nrm = fmaxf(sqrtf(red[0]), 1e-12f);
    dmap[(size_t)pix * DDIM + d] = v / nrm;
}

// ------------------------------- bilinear sample + final L2 normalize (NHWC dmap)
__global__ __launch_bounds__(256)
void sample_kernel(const float* __restrict__ dmap, const float* __restrict__ coords,
                   float* __restrict__ outSamp, int Hc, int Wc) {
    int b = blockIdx.x / KTOP;
    int k = blockIdx.x % KTOP;
    int d = threadIdx.x;
    float x = coords[((size_t)b * KTOP + k) * 2 + 0];
    float y = coords[((size_t)b * KTOP + k) * 2 + 1];
    float ix = x * 0.125f - 0.5f, iy = y * 0.125f - 0.5f;
    float fx = floorf(ix), fy = floorf(iy);
    float dx = ix - fx, dy = iy - fy;
    int x0 = (int)fx, y0 = (int)fy;
    float acc = 0.f;
    auto g = [&](int xi, int yi, float w) {
        bool valid = (xi >= 0 && xi < Wc && yi >= 0 && yi < Hc);
        int xc = min(max(xi, 0), Wc - 1), yc = min(max(yi, 0), Hc - 1);
        acc += dmap[(((size_t)b * Hc + yc) * Wc + xc) * DDIM + d] * (valid ? w : 0.f);
    };
    g(x0, y0, (1.f - dx) * (1.f - dy));
    g(x0 + 1, y0, dx * (1.f - dy));
    g(x0, y0 + 1, (1.f - dx) * dy);
    g(x0 + 1, y0 + 1, dx * dy);
    __shared__ float red[256];
    red[d] = acc * acc;
    __syncthreads();
    for (int s = 128; s > 0; s >>= 1) { if (d < s) red[d] += red[d + s]; __syncthreads(); }
    float nrm = fmaxf(sqrtf(red[0]), 1e-12f);
    outSamp[((size_t)b * DDIM + d) * KTOP + k] = acc / nrm;
}

// ================================================================= launcher
extern "C" void kernel_launch(void* const* d_in, const int* in_sizes, int n_in,
                              void* d_out, int out_size, void* d_ws, size_t ws_size,
                              hipStream_t stream) {
    (void)in_sizes; (void)n_in; (void)out_size; (void)ws_size;
    const float* image = (const float*)d_in[0];
    const float* wF[12]; const float* bF[12];
    for (int l = 0; l < 12; ++l) { wF[l] = (const float*)d_in[1 + 2 * l]; bF[l] = (const float*)d_in[2 + 2 * l]; }
    // 0:w1a 1:w1b 2:w2a 3:w2b 4:w3a 5:w3b 6:w4a 7:w4b 8:wPa 9:wPb 10:wDa 11:wDb
    static const int wCo[12] = {64, 64, 64, 64, 128, 128, 128, 128, 256, 65, 256, 256};
    static const int wCi[12] = {1, 64, 64, 64, 64, 128, 128, 128, 128, 256, 128, 256};
    static const int wTp[12] = {9, 9, 9, 9, 9, 9, 9, 9, 9, 1, 9, 1};

    char* basep = (char*)d_ws;
    size_t off = 0;
    auto take = [&](size_t bytes) { off = (off + 255) & ~(size_t)255; size_t o = off; off += bytes; return (void*)(basep + o); };

    _Float16* wH[12];
    for (int l = 1; l < 12; ++l) wH[l] = (_Float16*)take((size_t)wCo[l] * wCi[l] * wTp[l] * 2);
    _Float16* PA = (_Float16*)take((size_t)BB * 64 * HH * WW * 2);
    _Float16* PB = (_Float16*)take((size_t)BB * 64 * HH * WW * 2);
    _Float16* PC = (_Float16*)take((size_t)BB * 256 * HC * WC * 2);
    const size_t NF = (size_t)BB * HH * WW;
    float* S0 = (float*)take(NF * 4);
    float* T  = (float*)take(NF * 4);
    float* S2 = (float*)take(NF * 4);
    float* S3 = (float*)take(NF * 4);
    float* S4 = (float*)take(NF * 4);
    float* M  = (float*)take(NF * 4);
    float* S1 = (float*)take(NF * 4);
    float* DMAP = (float*)take((size_t)BB * DDIM * HC * WC * 4);

    float* outCoord = (float*)d_out;
    float* outScore = outCoord + (size_t)BB * KTOP * 2;
    float* outSamp  = outScore + (size_t)BB * KTOP;

    for (int l = 1; l < 12; ++l) {
        int n = wCo[l] * wCi[l] * wTp[l];
        cvt_w_kernel<<<(n + 255) / 256, 256, 0, stream>>>(wF[l], wH[l], wCo[l], wCi[l], wTp[l]);
    }

    conv1a_kernel<<<(BB * HH * WW + 255) / 256, 256, 0, stream>>>(image, wF[0], bF[0], PA, HH, WW, BB);

    auto conv3 = [&](const _Float16* in, int l, _Float16* out, int Cin, int Cout, int H, int W, int relu) {
        dim3 grid(W >> 3, ((H + 7) >> 3) * BB, (Cout + 63) / 64);
        conv_wmma_kernel<3><<<grid, 256, 0, stream>>>(in, wH[l], bF[l], out, Cin, Cout, H, W, relu);
    };
    auto conv1 = [&](const _Float16* in, int l, _Float16* out, int Cin, int Cout, int H, int W, int relu) {
        dim3 grid(W >> 3, ((H + 7) >> 3) * BB, (Cout + 63) / 64);
        conv_wmma_kernel<1><<<grid, 256, 0, stream>>>(in, wH[l], bF[l], out, Cin, Cout, H, W, relu);
    };
    auto pool = [&](const _Float16* in, _Float16* out, int C, int H, int W) {
        int n = BB * C * (H >> 1) * (W >> 1);
        pool2_kernel<<<(n + 255) / 256, 256, 0, stream>>>(in, out, C, H, W, BB);
    };

    conv3(PA, 1, PB, 64, 64, 480, 640, 1);     // conv1b
    pool(PB, PA, 64, 480, 640);
    conv3(PA, 2, PB, 64, 64, 240, 320, 1);     // conv2a
    conv3(PB, 3, PA, 64, 64, 240, 320, 1);     // conv2b
    pool(PA, PB, 64, 240, 320);
    conv3(PB, 4, PA, 64, 128, 120, 160, 1);    // conv3a
    conv3(PA, 5, PB, 128, 128, 120, 160, 1);   // conv3b
    pool(PB, PA, 128, 120, 160);
    conv3(PA, 6, PB, 128, 128, 60, 80, 1);     // conv4a
    conv3(PB, 7, PA, 128, 128, 60, 80, 1);     // conv4b -> x in PA

    // detector head
    conv3(PA, 8, PB, 128, 256, 60, 80, 1);     // convPa
    conv1(PB, 9, PC, 256, 65, 60, 80, 0);      // convPb
    softmax_shuffle_kernel<<<(BB * HC * WC + 255) / 256, 256, 0, stream>>>(PC, S0, HC, WC, BB);

    const int nfull = (int)NF;
    dim3 gF((nfull + 255) / 256);
    mp9h_kernel<<<gF, 256, 0, stream>>>(S0, T, HH, WW, BB);
    mp9v_kernel<<<gF, 256, 0, stream>>>(T, S2, HH, WW, BB);
    eqmask_kernel<<<gF, 256, 0, stream>>>(S0, S2, M, nfull);
    for (int it = 0; it < 2; ++it) {
        mp9h_kernel<<<gF, 256, 0, stream>>>(M, T, HH, WW, BB);
        mp9v_kernel<<<gF, 256, 0, stream>>>(T, S2, HH, WW, BB);
        suppress_kernel<<<gF, 256, 0, stream>>>(S0, S2, S3, nfull);
        mp9h_kernel<<<gF, 256, 0, stream>>>(S3, T, HH, WW, BB);
        mp9v_kernel<<<gF, 256, 0, stream>>>(T, S4, HH, WW, BB);
        update_kernel<<<gF, 256, 0, stream>>>(M, S3, S4, S2, nfull);
    }
    finalmask_kernel<<<gF, 256, 0, stream>>>(S0, M, S1, nfull);
    topk_kernel<<<BB, 1024, 0, stream>>>(S1, outCoord, outScore, HH, WW);

    // descriptor head
    conv3(PA, 10, PB, 128, 256, 60, 80, 1);    // convDa
    conv1(PB, 11, PC, 256, 256, 60, 80, 0);    // convDb
    l2norm_map_kernel<<<BB * HC * WC, 256, 0, stream>>>(PC, DMAP, HC, WC);
    sample_kernel<<<BB * KTOP, 256, 0, stream>>>(DMAP, outCoord, outSamp, HC, WC);
}
